// WindowMSA_49546742726934
// MI455X (gfx1250) — compile-verified, matrix operand order checked by reading, not run on hardware
//
#include <hip/hip_runtime.h>
#include <hip/hip_bf16.h>

typedef __attribute__((ext_vector_type(16))) _Float16 v16h;
typedef __attribute__((ext_vector_type(8)))  _Float16 v8h;
typedef __attribute__((ext_vector_type(8)))  float    v8f;

#define C_DIM 256
#define HW 224
#define NWIN 28
#define TOK 64
#define HD 32
#define WPB 2              // windows per block (M = 128)

// LDS layout (bytes) — total 327680 (= CDNA5 max 320KB/workgroup)
#define XS_OFF 0u          // [128][256] f16 : X, later attn-out O           (64KB)
#define QK_OFF 65536u      // [128][512] f16 : Q|K ; reused as f32 proj-out  (128KB)
#define VT_OFF 196608u     // [2][256][64] f16 : V transposed per window     (64KB)
#define SS_OFF 262144u     // 8 waves x [64][64] f16 scores/probs            (64KB)
#define SMEM_BYTES 327680u

// WMMA operand fragment load (16-bit A layout; B identical with row==N).
// Row-major source, ld in halves, 16B-aligned columns.
__device__ __forceinline__ v16h load_frag(const _Float16* base, int ld,
                                          int row_base, int col_base, int lane) {
    int r  = lane & 15;
    int hi = lane >> 4;
    const _Float16* p = base + (row_base + r) * ld + col_base + hi * 8;
    v8h lo = *(const v8h*)p;         // K = hi*8 .. +7
    v8h hv = *(const v8h*)(p + 16);  // K = 16 + hi*8 .. +7
    v16h out;
#pragma unroll
    for (int i = 0; i < 8; ++i) { out[i] = lo[i]; out[i + 8] = hv[i]; }
    return out;
}

// 16x16 f32 accumulator tile store (C/D layout: M = r + 8*(lane>>4), N = lane&15)
__device__ __forceinline__ void store_tile_f16(_Float16* base, int ld, int row_base,
                                               int col_base, int lane, const v8f& acc) {
    int n = lane & 15, hi = lane >> 4;
#pragma unroll
    for (int r = 0; r < 8; ++r)
        base[(row_base + r + 8 * hi) * ld + col_base + n] = (_Float16)acc[r];
}

__device__ __forceinline__ void store_tile_f32(float* base, int ld, int row_base,
                                               int col_base, int lane, const v8f& acc) {
    int n = lane & 15, hi = lane >> 4;
#pragma unroll
    for (int r = 0; r < 8; ++r)
        base[(row_base + r + 8 * hi) * ld + col_base + n] = acc[r];
}

// V tile store, transposed into per-window Vt[w][vcol][token]; rows m span 2 windows
__device__ __forceinline__ void store_vt(_Float16* vt, int mt, int vcol_base,
                                         int lane, const v8f& acc) {
    int n = lane & 15, hi = lane >> 4;
#pragma unroll
    for (int r = 0; r < 8; ++r) {
        int m = mt * 16 + r + 8 * hi;          // 0..127
        int w = m >> 6, t = m & 63;
        vt[w * (256 * TOK) + (vcol_base + n) * TOK + t] = (_Float16)acc[r];
    }
}

__global__ void convert_weights_kernel(const float* __restrict__ wqkv,
                                       const float* __restrict__ wproj,
                                       _Float16* __restrict__ dst) {
    int i = blockIdx.x * blockDim.x + threadIdx.x;
    const int NQ = 768 * C_DIM, NP = C_DIM * C_DIM;
    if (i < NQ)            dst[i] = (_Float16)wqkv[i];
    else if (i < NQ + NP)  dst[i] = (_Float16)wproj[i - NQ];
}

__global__ __launch_bounds__(256) void win_msa_kernel(
    const float* __restrict__ x, const float* __restrict__ b_proj,
    const _Float16* __restrict__ wqkv_h, const _Float16* __restrict__ wproj_h,
    float* __restrict__ out) {
    extern __shared__ char smem[];
    _Float16* Xs  = (_Float16*)(smem + XS_OFF);   // [128][256]
    _Float16* QKs = (_Float16*)(smem + QK_OFF);   // [128][512]
    _Float16* Vts = (_Float16*)(smem + VT_OFF);   // [2][256][64]
    _Float16* Ss  = (_Float16*)(smem + SS_OFF);   // [8][64][64]

    const int tid = threadIdx.x, lane = tid & 31, wv = tid >> 5;

    // coordinates of the two windows handled by this block
    int ba[WPB], ya[WPB], xa[WPB];
#pragma unroll
    for (int w = 0; w < WPB; ++w) {
        int wid = blockIdx.x * WPB + w;
        int b  = wid / (NWIN * NWIN);
        int rm = wid % (NWIN * NWIN);
        ba[w] = b; ya[w] = (rm / NWIN) * 8; xa[w] = (rm % NWIN) * 8;
    }

    if (tid == 0) __builtin_prefetch(wqkv_h, 0, 3);

    // ---- Phase 0: 2 windows -> LDS f16, float4 loads (16B-aligned x-rows) ----
#pragma unroll 4
    for (int i = 0; i < 32; ++i) {
        int idx = (i * 256 + tid) * 4;                 // element index
        int t2 = idx & 127, c = idx >> 7;
        int w = t2 >> 6, t = t2 & 63;
        const float* src =
            &x[((ba[w] * C_DIM + c) * HW + ya[w] + (t >> 3)) * HW + xa[w] + (t & 7)];
        float4 v4 = *(const float4*)src;               // global_load_b128
        Xs[(t2 + 0) * C_DIM + c] = (_Float16)v4.x;
        Xs[(t2 + 1) * C_DIM + c] = (_Float16)v4.y;
        Xs[(t2 + 2) * C_DIM + c] = (_Float16)v4.z;
        Xs[(t2 + 3) * C_DIM + c] = (_Float16)v4.w;
    }
    __syncthreads();

    // ---- Phase 1: QKV GEMM  qkv(128x768) = X(128x256) @ Wqkv^T ----
    for (int j = 0; j < 6; ++j) {
        int nt = wv * 6 + j;                 // output features nt*16..+15
        v8f accs[8];
#pragma unroll
        for (int mt = 0; mt < 8; ++mt) accs[mt] = (v8f)0.f;
        for (int kk = 0; kk < 8; ++kk) {
            v16h bf = load_frag(wqkv_h, C_DIM, nt * 16, kk * 32, lane);
            v16h af[8];
#pragma unroll
            for (int mt = 0; mt < 8; ++mt)           // one ds clause, one wait
                af[mt] = load_frag(Xs, C_DIM, mt * 16, kk * 32, lane);
#pragma unroll
            for (int mt = 0; mt < 8; ++mt)           // 8 back-to-back WMMAs
                accs[mt] = __builtin_amdgcn_wmma_f32_16x16x32_f16(
                    false, af[mt], false, bf, (short)0, accs[mt], false, false);
        }
        int db = nt * 16;
#pragma unroll
        for (int mt = 0; mt < 8; ++mt) {
            if (db < 512) store_tile_f16(QKs, 512, mt * 16, db, lane, accs[mt]);
            else          store_vt(Vts, mt, db - 512, lane, accs[mt]);
        }
    }
    __syncthreads();

    // ---- Phase 2: attention; wave wv = head wv, both windows sequentially ----
    {
        _Float16* Sw = Ss + wv * (TOK * TOK);          // private 64x64 f16
        for (int w = 0; w < WPB; ++w) {
            const int rb = w * TOK;
            v16h kf[4];
#pragma unroll
            for (int nt = 0; nt < 4; ++nt)
                kf[nt] = load_frag(QKs, 512, rb + nt * 16, 256 + wv * HD, lane);
#pragma unroll
            for (int mt = 0; mt < 4; ++mt) {
                v16h qf = load_frag(QKs, 512, rb + mt * 16, wv * HD, lane);
#pragma unroll
                for (int nt = 0; nt < 4; ++nt) {
                    v8f s = (v8f)0.f;
                    s = __builtin_amdgcn_wmma_f32_16x16x32_f16(
                        false, qf, false, kf[nt], (short)0, s, false, false);
                    int n = lane & 15, hi = lane >> 4;
#pragma unroll
                    for (int r = 0; r < 8; ++r)
                        Sw[(mt * 16 + r + 8 * hi) * TOK + nt * 16 + n] =
                            (_Float16)(s[r] * 0.1767766952966369f);
                }
            }
            __syncthreads();   // uniform: all waves loop w identically

            // row softmax: 2 rows/lane, f16 in place
#pragma unroll
            for (int rr = 0; rr < 2; ++rr) {
                _Float16* row = Sw + (lane + rr * 32) * TOK;
                float mx = -3.0e38f;
                for (int k2 = 0; k2 < TOK; ++k2) mx = fmaxf(mx, (float)row[k2]);
                float sum = 0.f;
                for (int k2 = 0; k2 < TOK; ++k2) {
                    float e = __expf((float)row[k2] - mx);
                    sum += e; row[k2] = (_Float16)e;
                }
                float inv = 1.0f / sum;
                for (int k2 = 0; k2 < TOK; ++k2)
                    row[k2] = (_Float16)((float)row[k2] * inv);
            }
            __syncthreads();

            // O_h(64x32) = P(64x64) @ V_h ; B operand = Vt rows (hd x token)
            const _Float16* Vw = Vts + w * (256 * TOK);
            v16h vf[2][2];
#pragma unroll
            for (int nt = 0; nt < 2; ++nt)
#pragma unroll
                for (int kk = 0; kk < 2; ++kk)
                    vf[nt][kk] = load_frag(Vw, TOK, wv * HD + nt * 16, kk * 32, lane);
#pragma unroll
            for (int mt = 0; mt < 4; ++mt) {
                v16h pf0 = load_frag(Sw, TOK, mt * 16, 0, lane);
                v16h pf1 = load_frag(Sw, TOK, mt * 16, 32, lane);
#pragma unroll
                for (int nt = 0; nt < 2; ++nt) {
                    v8f o = (v8f)0.f;
                    o = __builtin_amdgcn_wmma_f32_16x16x32_f16(
                        false, pf0, false, vf[nt][0], (short)0, o, false, false);
                    o = __builtin_amdgcn_wmma_f32_16x16x32_f16(
                        false, pf1, false, vf[nt][1], (short)0, o, false, false);
                    store_tile_f16(Xs, C_DIM, rb + mt * 16, wv * HD + nt * 16, lane, o);
                }
            }
            __syncthreads();
        }
    }

    // ---- Phase 3: proj GEMM  Y(128x256) = O(128x256) @ Wproj^T + b ----
    float* Pout = (float*)(smem + QK_OFF);    // [128][256] f32, reuses QK region
#pragma unroll
    for (int j = 0; j < 2; ++j) {
        int nt = wv * 2 + j;
        float bias = b_proj[nt * 16 + (lane & 15)];   // tile column bias
        v8f accs[8];
#pragma unroll
        for (int mt = 0; mt < 8; ++mt) accs[mt] = (v8f)0.f;
        for (int kk = 0; kk < 8; ++kk) {
            v16h bf = load_frag(wproj_h, C_DIM, nt * 16, kk * 32, lane);
            v16h af[8];
#pragma unroll
            for (int mt = 0; mt < 8; ++mt)
                af[mt] = load_frag(Xs, C_DIM, mt * 16, kk * 32, lane);
#pragma unroll
            for (int mt = 0; mt < 8; ++mt)
                accs[mt] = __builtin_amdgcn_wmma_f32_16x16x32_f16(
                    false, af[mt], false, bf, (short)0, accs[mt], false, false);
        }
#pragma unroll
        for (int mt = 0; mt < 8; ++mt) {
            accs[mt] = accs[mt] + bias;               // fold bias here
            store_tile_f32(Pout, C_DIM, mt * 16, nt * 16, lane, accs[mt]);
        }
    }
    __syncthreads();

    // ---- writeout: pure LDS->global copy, float4 stores (16B-aligned) ----
#pragma unroll 4
    for (int i = 0; i < 32; ++i) {
        int idx = (i * 256 + tid) * 4;
        int t2 = idx & 127, c = idx >> 7;
        int w = t2 >> 6, t = t2 & 63;
        float4 v4;
        v4.x = Pout[(t2 + 0) * C_DIM + c];
        v4.y = Pout[(t2 + 1) * C_DIM + c];
        v4.z = Pout[(t2 + 2) * C_DIM + c];
        v4.w = Pout[(t2 + 3) * C_DIM + c];
        float* dst =
            &out[((ba[w] * C_DIM + c) * HW + ya[w] + (t >> 3)) * HW + xa[w] + (t & 7)];
        *(float4*)dst = v4;                            // global_store_b128
    }
}

extern "C" void kernel_launch(void* const* d_in, const int* in_sizes, int n_in,
                              void* d_out, int out_size, void* d_ws, size_t ws_size,
                              hipStream_t stream) {
    const float* x      = (const float*)d_in[0];
    const float* w_qkv  = (const float*)d_in[1];
    const float* w_proj = (const float*)d_in[2];
    const float* b_proj = (const float*)d_in[3];
    float* out = (float*)d_out;

    _Float16* w16 = (_Float16*)d_ws;                 // 768*256 + 256*256 halves
    const _Float16* wqkv_h  = w16;
    const _Float16* wproj_h = w16 + 768 * C_DIM;

    (void)hipFuncSetAttribute((const void*)win_msa_kernel,
                              hipFuncAttributeMaxDynamicSharedMemorySize,
                              (int)SMEM_BYTES);

    convert_weights_kernel<<<(768 * C_DIM + C_DIM * C_DIM + 255) / 256, 256, 0, stream>>>(
        w_qkv, w_proj, w16);

    const int nblk = (4 * NWIN * NWIN) / WPB;   // 3136 windows / 2 = 1568 blocks
    win_msa_kernel<<<nblk, 256, SMEM_BYTES, stream>>>(x, b_proj, wqkv_h, wproj_h, out);
    (void)in_sizes; (void)n_in; (void)out_size; (void)ws_size;
}
